// Block_38474317037919
// MI455X (gfx1250) — matrix operand order repaired and checked by reading, not compiled, not run on hardware
//
#include <hip/hip_runtime.h>

typedef __attribute__((ext_vector_type(16))) __bf16 v16bf;
typedef __attribute__((ext_vector_type(8)))  __bf16 v8bf;
typedef __attribute__((ext_vector_type(8)))  float  v8f;

#define NH   12
#define DM   768
#define DFF  3072
#define DH   64
#define NB   4
#define NT   2048
#define MROWS (NB*NT)   // 8192

#define WAIT_ASYNC(n) asm volatile("s_wait_asynccnt " #n ::: "memory")

static __device__ inline v8f vzero8() { v8f z = {}; return z; }

static __device__ inline v16bf cat8(v8bf lo, v8bf hi) {
    return __builtin_shufflevector(lo, hi, 0,1,2,3,4,5,6,7,8,9,10,11,12,13,14,15);
}

// A-fragment (16x32 bf16, M x K): lane L holds row L%16; contiguous
// K 0..15 for lanes 0-15, 16..31 for lanes 16-31.
static __device__ inline v16bf load_a_frag(const __bf16* base, int stride, int lane) {
    const __bf16* p = base + (size_t)(lane & 15) * stride + ((lane < 16) ? 0 : 16);
    return cat8(*(const v8bf*)p, *(const v8bf*)(p + 8));
}

// B-fragment (32x16 bf16, K x N) sourced from row-major [N, K] storage:
// lane L holds column L%16; contiguous K 0..15 (lanes 0-15) / 16..31 (lanes 16-31).
static __device__ inline v16bf load_b_frag(const __bf16* base, int stride, int lane) {
    const __bf16* p = base + (size_t)(lane & 15) * stride + ((lane < 16) ? 0 : 16);
    return cat8(*(const v8bf*)p, *(const v8bf*)(p + 8));
}

static __device__ inline v8f wmma_bf16(v16bf a, v16bf b, v8f c) {
    return __builtin_amdgcn_wmma_f32_16x16x32_bf16(false, a, false, b, (short)0, c, false, false);
}

// Async DMA: 16 bytes per lane, global -> LDS, tracked by ASYNCcnt.
// Generic pointers to LDS carry the LDS byte offset in their low 32 bits.
static __device__ inline void async_copy16(const __bf16* gptr, const __bf16* lptr) {
    unsigned lds = (unsigned)(uintptr_t)lptr;
    asm volatile("global_load_async_to_lds_b128 %0, %1, off"
                 :: "v"(lds), "v"(gptr) : "memory");
}

// ---------------------------------------------------------------- conversion
__global__ void cvt_kernel(const float* __restrict__ in, __bf16* __restrict__ out, int n) {
    int i = blockIdx.x * 256 + threadIdx.x;
    if (i < n) out[i] = (__bf16)in[i];
}

// w_o [768(j), 768(c)] f32 -> out [c][j] bf16 so GEMM-B is [N,K] row-major
__global__ void transpose_kernel(const float* __restrict__ in, __bf16* __restrict__ out) {
    __shared__ float tile[16][17];
    int j0 = blockIdx.y * 16, c0 = blockIdx.x * 16;
    tile[threadIdx.y][threadIdx.x] = in[(size_t)(j0 + threadIdx.y) * DM + c0 + threadIdx.x];
    __syncthreads();
    out[(size_t)(c0 + threadIdx.y) * DM + j0 + threadIdx.x] = (__bf16)tile[threadIdx.x][threadIdx.y];
}

// ---------------------------------------------------------------- layernorm
__global__ __launch_bounds__(256)
void ln_kernel(const float* __restrict__ X, const float* __restrict__ g,
               const float* __restrict__ be, __bf16* __restrict__ out) {
    __shared__ float red[2][8];
    const int row = blockIdx.x;
    const float* x = X + (size_t)row * DM;
    const int t = threadIdx.x;
    float v0 = x[t], v1 = x[t + 256], v2 = x[t + 512];
    float s  = v0 + v1 + v2;
    float s2 = v0*v0 + v1*v1 + v2*v2;
    #pragma unroll
    for (int off = 16; off > 0; off >>= 1) {
        s  += __shfl_xor(s,  off);
        s2 += __shfl_xor(s2, off);
    }
    const int lane = t & 31, wave = t >> 5;
    if (lane == 0) { red[0][wave] = s; red[1][wave] = s2; }
    __syncthreads();
    float ts = 0.f, ts2 = 0.f;
    #pragma unroll
    for (int i = 0; i < 8; ++i) { ts += red[0][i]; ts2 += red[1][i]; }
    const float mu  = ts * (1.f / DM);
    const float var = ts2 * (1.f / DM) - mu * mu;
    const float r   = rsqrtf(var + 1e-5f);
    __bf16* o = out + (size_t)row * DM;
    o[t]       = (__bf16)((v0 - mu) * r * g[t]       + be[t]);
    o[t + 256] = (__bf16)((v1 - mu) * r * g[t + 256] + be[t + 256]);
    o[t + 512] = (__bf16)((v2 - mu) * r * g[t + 512] + be[t + 512]);
}

// ---------------------------------------------------------------- WMMA GEMM
// C[M,N] = A[M,K] * Bt[N,K]^T, bf16 inputs, f32 accumulate.
// Double-buffered async global->LDS pipeline, last iteration peeled so the
// steady-state loop is branch-free (unconditional issue + asynccnt<=8).
// mode 0: QKV scatter (Q scaled 1/8, V transposed)   mode 1: +resid -> f32
// mode 2: +bias, relu -> bf16                        mode 3: +bias +resid -> f32
__global__ __launch_bounds__(256)
void gemm_kernel(const __bf16* __restrict__ A, const __bf16* __restrict__ Bt,
                 int N, int K, int mode,
                 const float* __restrict__ bias, const float* __restrict__ resid,
                 float* __restrict__ outF, __bf16* __restrict__ outH,
                 __bf16* __restrict__ qOut, __bf16* __restrict__ kOut,
                 __bf16* __restrict__ vtOut) {
    __shared__ __attribute__((aligned(16))) __bf16 As[2][128][72];
    __shared__ __attribute__((aligned(16))) __bf16 Bs[2][128][72];

    const int tid  = threadIdx.x;
    const int lane = tid & 31, wave = tid >> 5;
    const int wm = wave & 3, wn = wave >> 2;           // 4 x 2 wave grid
    const int bm = blockIdx.y * 128, bn = blockIdx.x * 128;

    v8f acc[2][4];
    #pragma unroll
    for (int mi = 0; mi < 2; ++mi)
        #pragma unroll
        for (int ni = 0; ni < 4; ++ni) acc[mi][ni] = vzero8();

    const int ldrow = tid >> 1;          // 0..127
    const int ldcol = (tid & 1) * 32;    // 0 / 32

    auto issue_tile = [&](int buf, int k0) {
        const __bf16* ag = A  + (size_t)(bm + ldrow) * K + k0 + ldcol;
        const __bf16* bg = Bt + (size_t)(bn + ldrow) * K + k0 + ldcol;
        #pragma unroll
        for (int c = 0; c < 4; ++c) {
            async_copy16(ag + c * 8, &As[buf][ldrow][ldcol + c * 8]);
            async_copy16(bg + c * 8, &Bs[buf][ldrow][ldcol + c * 8]);
        }
    };
    auto compute_tile = [&](int buf) {
        #pragma unroll
        for (int ks = 0; ks < 64; ks += 32) {
            v16bf af[2], bf[4];
            #pragma unroll
            for (int mi = 0; mi < 2; ++mi)
                af[mi] = load_a_frag(&As[buf][wm * 32 + mi * 16][ks], 72, lane);
            #pragma unroll
            for (int ni = 0; ni < 4; ++ni)
                bf[ni] = load_b_frag(&Bs[buf][wn * 64 + ni * 16][ks], 72, lane);
            #pragma unroll
            for (int mi = 0; mi < 2; ++mi)
                #pragma unroll
                for (int ni = 0; ni < 4; ++ni)
                    acc[mi][ni] = wmma_bf16(af[mi], bf[ni], acc[mi][ni]);
        }
    };

    const int nk = K / 64;
    issue_tile(0, 0);
    int cur = 0;
    for (int kb = 0; kb < nk - 1; ++kb) {         // steady state: branch-free
        issue_tile(cur ^ 1, (kb + 1) * 64);
        WAIT_ASYNC(8);      // in-order completion: current tile's 8 copies done
        __syncthreads();
        compute_tile(cur);
        __syncthreads();    // all waves done reading before overwrite
        cur ^= 1;
    }
    WAIT_ASYNC(0);                                // peeled last tile
    __syncthreads();
    compute_tile(cur);

    #pragma unroll
    for (int mi = 0; mi < 2; ++mi) {
        #pragma unroll
        for (int ni = 0; ni < 4; ++ni) {
            const int row0 = bm + wm * 32 + mi * 16 + ((lane < 16) ? 0 : 8);
            const int col  = bn + wn * 64 + ni * 16 + (lane & 15);
            #pragma unroll
            for (int v = 0; v < 8; ++v) {
                const int row = row0 + v;
                float val = acc[mi][ni][v];
                if (mode == 0) {
                    const int b = row >> 11, t = row & 2047;
                    if (col < DM) {
                        const int h = col >> 6, d = col & 63;
                        qOut[(((size_t)b * NH + h) * NT + t) * DH + d] = (__bf16)(val * 0.125f);
                    } else if (col < 2 * DM) {
                        const int c = col - DM, h = c >> 6, d = c & 63;
                        kOut[(((size_t)b * NH + h) * NT + t) * DH + d] = (__bf16)val;
                    } else {
                        const int c = col - 2 * DM, h = c >> 6, d = c & 63;
                        vtOut[(((size_t)b * NH + h) * DH + d) * NT + t] = (__bf16)val;
                    }
                } else if (mode == 1) {
                    outF[(size_t)row * N + col] = resid[(size_t)row * N + col] + val;
                } else if (mode == 2) {
                    float x = val + bias[col];
                    outH[(size_t)row * N + col] = (__bf16)(x > 0.f ? x : 0.f);
                } else {
                    outF[(size_t)row * N + col] =
                        resid[(size_t)row * N + col] + val + bias[col];
                }
            }
        }
    }
}

// ---------------------------------------------------------------- attention
// Flash-style causal attention. grid (B*H, T/64), 4 waves, 16 query rows/wave.
// K/V chunks staged cooperatively in LDS via async copies, double-buffered,
// last chunk peeled for a branch-free steady-state loop.
__global__ __launch_bounds__(128)
void attn_kernel(const __bf16* __restrict__ Q, const __bf16* __restrict__ Km,
                 const __bf16* __restrict__ Vt, __bf16* __restrict__ Y) {
    __shared__ __attribute__((aligned(16))) __bf16 Ks[2][32][72];
    __shared__ __attribute__((aligned(16))) __bf16 Vs[2][64][40];
    __shared__ __attribute__((aligned(16))) __bf16 P[4][16][40];
    const int tid = threadIdx.x;
    const int lane = tid & 31, wave = tid >> 5;
    const int bh = blockIdx.x;
    const int b = bh / NH, h = bh % NH;
    const int qblock = blockIdx.y;
    const int q0 = qblock * 64 + wave * 16;
    const __bf16* qbase = Q  + (size_t)bh * NT * DH;
    const __bf16* kbase = Km + (size_t)bh * NT * DH;
    const __bf16* vbase = Vt + (size_t)bh * DH * NT;

    // resident Q fragments (K-contraction dh = 0..31, 32..63); Q pre-scaled by 1/8
    v16bf qf[2];
    {
        const __bf16* p = qbase + (size_t)(q0 + (lane & 15)) * DH;
        const int off = (lane < 16) ? 0 : 16;
        qf[0] = cat8(*(const v8bf*)(p + off),      *(const v8bf*)(p + off + 8));
        qf[1] = cat8(*(const v8bf*)(p + 32 + off), *(const v8bf*)(p + 32 + off + 8));
    }

    v8f o[4];
    #pragma unroll
    for (int ni = 0; ni < 4; ++ni) o[ni] = vzero8();
    float mrow[8], lrow[8];
    #pragma unroll
    for (int v = 0; v < 8; ++v) { mrow[v] = -1e30f; lrow[v] = 0.f; }

    // cooperative async staging of one 32-key chunk of K and V
    const int krow = tid >> 2, kseg = (tid & 3) * 16;   // K: 32 rows x 64 dh
    const int vrow = tid >> 1, vseg = (tid & 1) * 16;   // V: 64 rows x 32 t
    auto issue_chunk = [&](int buf, int s0) {
        const __bf16* kg = kbase + (size_t)(s0 + krow) * DH + kseg;
        async_copy16(kg,     &Ks[buf][krow][kseg]);
        async_copy16(kg + 8, &Ks[buf][krow][kseg + 8]);
        const __bf16* vg = vbase + (size_t)vrow * NT + s0 + vseg;
        async_copy16(vg,     &Vs[buf][vrow][vseg]);
        async_copy16(vg + 8, &Vs[buf][vrow][vseg + 8]);
    };

    const int half8   = (lane < 16) ? 0 : 8;
    const int boff    = (lane < 16) ? 0 : 16;
    const int aoff    = (lane < 16) ? 0 : 16;

    auto process_chunk = [&](int cur, int s0) {
        // ---- scores: two 16x16 tiles, contraction over DH=64 (from LDS)
        v8f s[2];
        #pragma unroll
        for (int nt = 0; nt < 2; ++nt) {
            s[nt] = vzero8();
            const __bf16* kp = &Ks[cur][nt * 16 + (lane & 15)][0];
            v16bf k0f = cat8(*(const v8bf*)(kp + boff),      *(const v8bf*)(kp + boff + 8));
            v16bf k1f = cat8(*(const v8bf*)(kp + 32 + boff), *(const v8bf*)(kp + 32 + boff + 8));
            s[nt] = wmma_bf16(qf[0], k0f, s[nt]);
            s[nt] = wmma_bf16(qf[1], k1f, s[nt]);
        }
        // ---- online softmax update (per row = (v, half-wave))
        const int col0 = s0 + (lane & 15);
        #pragma unroll
        for (int v = 0; v < 8; ++v) {
            const int qrow = q0 + v + half8;
            float s0v = (col0      > qrow) ? -1e30f : s[0][v];
            float s1v = (col0 + 16 > qrow) ? -1e30f : s[1][v];
            float t = fmaxf(s0v, s1v);
            t = fmaxf(t, __shfl_xor(t, 1));
            t = fmaxf(t, __shfl_xor(t, 2));
            t = fmaxf(t, __shfl_xor(t, 4));
            t = fmaxf(t, __shfl_xor(t, 8));
            const float mnew  = fmaxf(mrow[v], t);
            const float scale = __expf(mrow[v] - mnew);
            const float p0 = __expf(s0v - mnew);
            const float p1 = __expf(s1v - mnew);
            float ps = p0 + p1;
            ps += __shfl_xor(ps, 1);
            ps += __shfl_xor(ps, 2);
            ps += __shfl_xor(ps, 4);
            ps += __shfl_xor(ps, 8);
            lrow[v] = lrow[v] * scale + ps;
            mrow[v] = mnew;
            #pragma unroll
            for (int ni = 0; ni < 4; ++ni) o[ni][v] *= scale;
            const int prow = v + half8;
            P[wave][prow][lane & 15]        = (__bf16)p0;
            P[wave][prow][16 + (lane & 15)] = (__bf16)p1;
        }
        asm volatile("s_wait_dscnt 0" ::: "memory");   // P tile is wave-private
        // ---- P(16x32, A-layout from LDS) x V(32x16 per d-tile, from LDS)
        const __bf16* pp = &P[wave][lane & 15][0];
        v16bf pf = cat8(*(const v8bf*)(pp + aoff), *(const v8bf*)(pp + aoff + 8));
        #pragma unroll
        for (int ni = 0; ni < 4; ++ni) {
            const __bf16* vp = &Vs[cur][ni * 16 + (lane & 15)][boff];
            v16bf vf = cat8(*(const v8bf*)vp, *(const v8bf*)(vp + 8));
            o[ni] = wmma_bf16(pf, vf, o[ni]);
        }
    };

    const int nchunks = qblock * 2 + 2;      // causal pruning: keys <= qblock end
    issue_chunk(0, 0);
    int cur = 0;
    for (int c = 0; c < nchunks - 1; ++c) {  // steady state: branch-free
        issue_chunk(cur ^ 1, (c + 1) * 32);
        WAIT_ASYNC(4);
        __syncthreads();
        process_chunk(cur, c * 32);
        __syncthreads();                     // done reading Ks/Vs before overwrite
        cur ^= 1;
    }
    WAIT_ASYNC(0);                           // peeled last chunk
    __syncthreads();
    process_chunk(cur, (nchunks - 1) * 32);

    // ---- normalize + write Y [B*T, 768] bf16
    #pragma unroll
    for (int ni = 0; ni < 4; ++ni) {
        const int d = ni * 16 + (lane & 15);
        #pragma unroll
        for (int v = 0; v < 8; ++v) {
            const int t = q0 + v + half8;
            const float y = o[ni][v] / lrow[v];
            Y[((size_t)(b * NT + t)) * DM + h * DH + d] = (__bf16)y;
        }
    }
}

// ---------------------------------------------------------------- launcher
extern "C" void kernel_launch(void* const* d_in, const int* in_sizes, int n_in,
                              void* d_out, int out_size, void* d_ws, size_t ws_size,
                              hipStream_t stream) {
    const float* X   = (const float*)d_in[0];
    const float* wq  = (const float*)d_in[1];
    const float* wk  = (const float*)d_in[2];
    const float* wv  = (const float*)d_in[3];
    const float* wo  = (const float*)d_in[4];
    const float* W1  = (const float*)d_in[5];
    const float* b1  = (const float*)d_in[6];
    const float* W2  = (const float*)d_in[7];
    const float* b2  = (const float*)d_in[8];
    const float* g1  = (const float*)d_in[9];
    const float* be1 = (const float*)d_in[10];
    const float* g2  = (const float*)d_in[11];
    const float* be2 = (const float*)d_in[12];
    float* outF = (float*)d_out;

    char* ws = (char*)d_ws;
    size_t off = 0;
    auto alloc = [&](size_t bytes) -> void* {
        void* p = ws + off;
        off = (off + bytes + 255) & ~(size_t)255;
        return p;
    };
    __bf16* wqkv = (__bf16*)alloc((size_t)3 * DM * DM * 2);
    __bf16* woT  = (__bf16*)alloc((size_t)DM * DM * 2);
    __bf16* w1b  = (__bf16*)alloc((size_t)DFF * DM * 2);
    __bf16* w2b  = (__bf16*)alloc((size_t)DM * DFF * 2);
    __bf16* xn   = (__bf16*)alloc((size_t)MROWS * DM * 2);
    __bf16* qb   = (__bf16*)alloc((size_t)NB * NH * NT * DH * 2);
    __bf16* kb   = (__bf16*)alloc((size_t)NB * NH * NT * DH * 2);
    __bf16* vtb  = (__bf16*)alloc((size_t)NB * NH * NT * DH * 2);
    __bf16* yb   = (__bf16*)alloc((size_t)MROWS * DM * 2);
    __bf16* hb   = (__bf16*)alloc((size_t)MROWS * DFF * 2);

    const int nw = DM * DM;           // 589824
    const int nf = DFF * DM;          // 2359296
    cvt_kernel<<<(nw + 255) / 256, 256, 0, stream>>>(wq, wqkv,           nw);
    cvt_kernel<<<(nw + 255) / 256, 256, 0, stream>>>(wk, wqkv + nw,      nw);
    cvt_kernel<<<(nw + 255) / 256, 256, 0, stream>>>(wv, wqkv + 2 * nw,  nw);
    cvt_kernel<<<(nf + 255) / 256, 256, 0, stream>>>(W1, w1b,            nf);
    cvt_kernel<<<(nf + 255) / 256, 256, 0, stream>>>(W2, w2b,            nf);
    transpose_kernel<<<dim3(DM / 16, DM / 16), dim3(16, 16), 0, stream>>>(wo, woT);

    // LN1 -> xn (bf16)
    ln_kernel<<<MROWS, 256, 0, stream>>>(X, g1, be1, xn);
    // QKV projection: [8192,768] x [2304,768]^T, scatter to Q/K/Vt
    gemm_kernel<<<dim3(3 * DM / 128, MROWS / 128), 256, 0, stream>>>(
        xn, wqkv, 3 * DM, DM, 0, nullptr, nullptr, nullptr, nullptr, qb, kb, vtb);
    // causal flash attention -> yb
    attn_kernel<<<dim3(NB * NH, NT / 64), 128, 0, stream>>>(qb, kb, vtb, yb);
    // O-projection + residual -> X2 (fp32) in d_out
    gemm_kernel<<<dim3(DM / 128, MROWS / 128), 256, 0, stream>>>(
        yb, woT, DM, DM, 1, nullptr, X, outF, nullptr, nullptr, nullptr, nullptr);
    // LN2 -> xn (bf16)
    ln_kernel<<<MROWS, 256, 0, stream>>>(outF, g2, be2, xn);
    // FFN1 + bias + relu -> hb (bf16)
    gemm_kernel<<<dim3(DFF / 128, MROWS / 128), 256, 0, stream>>>(
        xn, w1b, DFF, DM, 2, b1, nullptr, nullptr, hb, nullptr, nullptr, nullptr);
    // FFN2 + bias + residual(X2) -> final fp32 in d_out
    gemm_kernel<<<dim3(DM / 128, MROWS / 128), 256, 0, stream>>>(
        hb, w2b, DM, DFF, 3, b2, outF, outF, nullptr, nullptr, nullptr, nullptr);
}
